// GAF_81990925680895
// MI455X (gfx1250) — matrix operand order repaired and checked by reading, not compiled, not run on hardware
//
#include <hip/hip_runtime.h>

// ---------------------------------------------------------------------------
// GAF (Autoformer/GMAN hybrid) forward pass for MI455X (gfx1250, wave32, WMMA)
//
// Shapes: B=16, L_HIS=L_PRED=12, N=325, D=64, DFF=128, heads 8x8, top_k=7.
// All dense linears -> v_wmma_f32_16x16x32_f16 (f16 in / f32 accumulate).
// GEMM templated on (N,K,MODE): static accumulators, branch-free epilogue,
// FULL bit elides all bounds checks when M % 64 == 0 (true for all big GEMMs).
// Spatial attention (inner dim 8) and autocorrelation (L=12) -> VALU + LDS.
// ---------------------------------------------------------------------------

typedef __attribute__((ext_vector_type(16))) _Float16 v16h;
typedef __attribute__((ext_vector_type(8)))  _Float16 v8h;
typedef __attribute__((ext_vector_type(8)))  float    v8f;

#define BN   16
#define LSEQ 12
#define NV   325
#define DM   64
#define DFF  128
#define TOK  (BN * LSEQ * NV)       // 62400 tokens (= 975 * 64)
#define CHN  (BN * NV * DM)         // 332800 (b,n,d) channels

// ============================ WMMA GEMM =====================================
// out[M x N] = act( A[M x K] @ W[N x K]^T [+ bias] [+ addsrc] ), N,K in {64,128}.
// MODE: bits[1:0] act (0 none, 1 relu, 2 sigmoid), bit2 hasBias, bit3 hasAdd,
//       bit4 M is a multiple of 64 (no bounds checks anywhere).
// Block: 256 threads (8 waves), 64 rows per block, whole K panel in LDS (f16).
// Wave w: row tile w>>1 (16 rows), column half w&1 (NCT = N/32 col tiles).
union Frag16 { v16h v; v8h h[2]; };

template <int N, int K, int MODE>
__global__ __launch_bounds__(256)
void k_gemm(const float* __restrict__ A, const float* __restrict__ W,
            const float* __restrict__ bias, const float* __restrict__ addsrc,
            float* __restrict__ out, int M)
{
    constexpr int  NCT  = (N >> 4) >> 1;  // col tiles per wave: 2 (N=64) / 4 (N=128)
    constexpr int  ACT  = MODE & 3;
    constexpr bool HB   = (MODE & 4) != 0;
    constexpr bool HA   = (MODE & 8) != 0;
    constexpr bool FULL = (MODE & 16) != 0;
    __shared__ alignas(32) _Float16 a_sh[64 * K];
    __shared__ alignas(32) _Float16 w_sh[N * K];

    const int tid = threadIdx.x;
    const int m0  = blockIdx.x * 64;

    for (int i = tid; i < 64 * K; i += 256) {
        int r = i / K, kk = i - r * K;        // K is constexpr -> shifts
        int gr = m0 + r;
        float v;
        if (FULL) v = A[(size_t)gr * K + kk];
        else      v = (gr < M) ? A[(size_t)gr * K + kk] : 0.0f;
        a_sh[r * K + kk] = (_Float16)v;
    }
    for (int i = tid; i < N * K; i += 256)
        w_sh[i] = (_Float16)W[i];
    __syncthreads();

    const int lane = tid & 31;
    const int wv   = tid >> 5;
    const int rt   = wv >> 1;          // row tile 0..3
    const int c0   = (wv & 1) * NCT;
    const int l16  = lane & 15;
    const int hi   = lane >> 4;

    v8f acc[NCT];
#pragma unroll
    for (int i = 0; i < NCT; ++i)
#pragma unroll
        for (int j = 0; j < 8; ++j) acc[i][j] = 0.0f;

#pragma unroll
    for (int k0 = 0; k0 < K; k0 += 32) {
        // A fragment (16x32 f16): lane<16 holds {K0..7,K16..23}, lane>=16 {K8..15,K24..31}
        Frag16 af;
        const int arow = (rt * 16 + l16) * K;
        af.h[0] = *(const v8h*)&a_sh[arow + k0 + 8 * hi];
        af.h[1] = *(const v8h*)&a_sh[arow + k0 + 16 + 8 * hi];
#pragma unroll
        for (int c = 0; c < NCT; ++c) {
            // B fragment (32x16): lane<16 holds W[n][{K0..7,K16..23}], lane>=16 {K8..15,K24..31}
            Frag16 bf;
            const int wrow = ((c0 + c) * 16 + l16) * K + k0 + 8 * hi;
            bf.h[0] = *(const v8h*)&w_sh[wrow];
            bf.h[1] = *(const v8h*)&w_sh[wrow + 16];
            acc[c] = __builtin_amdgcn_wmma_f32_16x16x32_f16(
                false, af.v, false, bf.v, (short)0, acc[c], false, false);
        }
    }

    const int row0 = m0 + rt * 16 + 8 * hi;   // first of this lane's 8 rows
#pragma unroll
    for (int c = 0; c < NCT; ++c) {
        const int col = (c0 + c) * 16 + l16;
        float bv;
        if (HB) bv = bias[col];
#pragma unroll
        for (int r = 0; r < 8; ++r) {
            const int row = row0 + r;         // C/D: VGPR r -> M=r / M=8+r
            if (FULL || row < M) {
                float x = acc[c][r];
                if (HB) x += bv;
                if (HA) x += addsrc[(size_t)row * N + col];
                if (ACT == 1)      x = x > 0.0f ? x : 0.0f;
                else if (ACT == 2) x = 1.0f / (1.0f + __expf(-x));
                out[(size_t)row * N + col] = x;
            }
        }
    }
}

// runtime -> template dispatch (only the MODE combos actually launched)
template <int N, int K>
static void launch_gemm_nk(const float* A, const float* W, const float* bias,
                           const float* add, float* out, int M, int act,
                           hipStream_t stream) {
    const int mode = act | (bias ? 4 : 0) | (add ? 8 : 0) | ((M % 64 == 0) ? 16 : 0);
    dim3 g((M + 63) / 64), blk(256);
#define GEMM_CASE(MD) case MD: \
    k_gemm<N, K, MD><<<g, blk, 0, stream>>>(A, W, bias, add, out, M); break;
    switch (mode) {
        GEMM_CASE(4)   // partial-M, bias            (ste small GEMMs)
        GEMM_CASE(5)   // partial-M, bias, relu      (ste s1)
        GEMM_CASE(16)  // full, plain                (gate xs)
        GEMM_CASE(17)  // full, relu                 (ff1)
        GEMM_CASE(20)  // full, bias                 (attn q/k/v, gate o2)
        GEMM_CASE(21)  // full, bias, relu           (sa q/k/v/o, gate o1, out o1)
        GEMM_CASE(24)  // full, add                  (ff2 + residual)
        GEMM_CASE(28)  // full, bias, add            (attn o + residual)
        GEMM_CASE(30)  // full, bias, add, sigmoid   (gate)
        default: break;
    }
#undef GEMM_CASE
}

// ======================= elementwise / embedding ============================
__global__ void k_embed(const float* __restrict__ X, const float* __restrict__ w,
                        const float* __restrict__ b, float* __restrict__ out) {
    int gid = blockIdx.x * blockDim.x + threadIdx.x;
    if (gid >= TOK * DM) return;
    int tok = gid >> 6, d = gid & 63;
    out[gid] = X[tok] * w[d] + b[d];
}

__global__ void k_te1(const int* __restrict__ TE, const float* __restrict__ t1w,
                      const float* __restrict__ t1b, float* __restrict__ out) {
    int gid = blockIdx.x * blockDim.x + threadIdx.x;
    if (gid >= BN * 24 * DM) return;
    int r = gid >> 6, d = gid & 63;
    int dow = TE[r * 2] % 7;
    int tod = TE[r * 2 + 1] % 288;
    float v = t1w[d * 295 + dow] + t1w[d * 295 + 7 + tod] + t1b[d];
    out[gid] = v > 0.0f ? v : 0.0f;
}

__global__ void k_ste(const float* __restrict__ se, const float* __restrict__ te2,
                      float* __restrict__ steh, float* __restrict__ step_) {
    int gid = blockIdx.x * blockDim.x + threadIdx.x;
    if (gid >= BN * 24 * NV * DM) return;
    int d = gid & 63, r = gid >> 6;
    int n = r % NV, q = r / NV;
    int l = q % 24, b = q / 24;
    float v = se[n * DM + d] + te2[(b * 24 + l) * DM + d];
    if (l < LSEQ) steh[(((size_t)(b * LSEQ + l) * NV + n) << 6) + d] = v;
    else          step_[(((size_t)(b * LSEQ + (l - LSEQ)) * NV + n) << 6) + d] = v;
}

__global__ void k_concat(const float* __restrict__ a, const float* __restrict__ b,
                         float* __restrict__ o) {
    int gid = blockIdx.x * blockDim.x + threadIdx.x;
    if (gid >= TOK * 128) return;
    int tok = gid >> 7, c = gid & 127;
    o[gid] = (c < 64) ? a[tok * 64 + c] : b[tok * 64 + (c - 64)];
}

__global__ void k_add(const float* __restrict__ a, const float* __restrict__ b,
                      float* __restrict__ o, int n) {
    int gid = blockIdx.x * blockDim.x + threadIdx.x;
    if (gid < n) o[gid] = a[gid] + b[gid];
}

__global__ void k_zero(float* __restrict__ p, int n) {
    int gid = blockIdx.x * blockDim.x + threadIdx.x;
    if (gid < n) p[gid] = 0.0f;
}

__global__ void k_gatecomb(const float* __restrict__ g, const float* __restrict__ hs,
                           const float* __restrict__ ht, float* __restrict__ o) {
    int gid = blockIdx.x * blockDim.x + threadIdx.x;
    if (gid >= TOK * DM) return;
    float gg = g[gid];
    o[gid] = gg * hs[gid] + (1.0f - gg) * ht[gid];
}

// ===================== spatial attention (8-dim heads) ======================
__global__ __launch_bounds__(352)
void k_sa_attn(const float* __restrict__ Q, const float* __restrict__ K,
               const float* __restrict__ V, float* __restrict__ O) {
    __shared__ float kls[NV * 8];
    __shared__ float vls[NV * 8];
    int blk = blockIdx.x;                 // B*12*8 = 1536 blocks
    int b = blk / 96, rem = blk % 96;
    int t = rem >> 3, c = rem & 7;
    const size_t base = ((size_t)(b * LSEQ + t) * NV) * DM + c * 8;
    for (int i = threadIdx.x; i < NV * 8; i += 352) {
        int m = i >> 3, j = i & 7;
        kls[i] = K[base + (size_t)m * DM + j];
        vls[i] = V[base + (size_t)m * DM + j];
    }
    __syncthreads();
    int n = threadIdx.x;
    if (n < NV) {
        float q[8];
        for (int j = 0; j < 8; ++j) q[j] = Q[base + (size_t)n * DM + j];
        float mx = -3.4e38f, Z = 0.0f, acc[8] = {0, 0, 0, 0, 0, 0, 0, 0};
        const float sc = 0.3535533905932738f;  // 1/sqrt(8)
        for (int m = 0; m < NV; ++m) {
            float s = 0.0f;
#pragma unroll
            for (int j = 0; j < 8; ++j) s += q[j] * kls[m * 8 + j];
            s *= sc;
            if (s > mx) {
                float cor = __expf(mx - s);
                Z *= cor;
#pragma unroll
                for (int j = 0; j < 8; ++j) acc[j] *= cor;
                mx = s;
            }
            float e = __expf(s - mx);
            Z += e;
#pragma unroll
            for (int j = 0; j < 8; ++j) acc[j] += e * vls[m * 8 + j];
        }
        float inv = 1.0f / Z;
        for (int j = 0; j < 8; ++j) O[base + (size_t)n * DM + j] = acc[j] * inv;
    }
}

// ============ autocorrelation core (circular corr, top-7, gather) ===========
__global__ __launch_bounds__(64)
void k_ac_core(const float* __restrict__ Q, const float* __restrict__ K,
               const float* __restrict__ V, float* __restrict__ O) {
    __shared__ float allc[64 * 12];
    __shared__ float mc[12];
    __shared__ float wsm[8];
    __shared__ int   dsm[8];
    int d = threadIdx.x;
    int b = blockIdx.x / NV, n = blockIdx.x % NV;
    float qv[12], kv[12], vv[12];
    for (int l = 0; l < 12; ++l) {
        size_t idx = (((size_t)(b * LSEQ + l) * NV + n) << 6) + d;
        qv[l] = Q[idx]; kv[l] = K[idx]; vv[l] = V[idx];
    }
    // corr[tau] = sum_l q[(l+tau)%12] * k[l]  (== irfft(Qf*conj(Kf)))
    for (int t = 0; t < 12; ++t) {
        float s = 0.0f;
        for (int l = 0; l < 12; ++l) s += qv[(l + t) % 12] * kv[l];
        allc[d * 12 + t] = s;
    }
    __syncthreads();
    if (d < 12) {
        float s = 0.0f;
        for (int j = 0; j < 64; ++j) s += allc[j * 12 + d];
        mc[d] = s * (1.0f / 64.0f);
    }
    __syncthreads();
    if (d == 0) {
        float a[12];
        for (int i = 0; i < 12; ++i) a[i] = mc[i];
        float w[7]; int id[7];
        for (int k = 0; k < 7; ++k) {           // stable top-7 (ties: lowest idx)
            int bi = 0; float bv = a[0];
            for (int i = 1; i < 12; ++i) if (a[i] > bv) { bv = a[i]; bi = i; }
            w[k] = bv; id[k] = bi; a[bi] = -3.4e38f;
        }
        float mxw = w[0], ss = 0.0f;
        for (int k = 0; k < 7; ++k) { float e = __expf(w[k] - mxw); wsm[k] = e; ss += e; }
        float inv = 1.0f / ss;
        for (int k = 0; k < 7; ++k) { wsm[k] *= inv; dsm[k] = id[k]; }
    }
    __syncthreads();
    for (int l = 0; l < 12; ++l) {
        float s = 0.0f;
        for (int k = 0; k < 7; ++k) s += wsm[k] * vv[(l + dsm[k]) % 12];
        O[(((size_t)(b * LSEQ + l) * NV + n) << 6) + d] = s;
    }
}

// ================= series decomposition (k=25 over L=12) ====================
// m[l] = (S + (12-l)*x0 + (l+1)*x11)/25 ; seasonal = x - m ; ta += m
__global__ void k_decomp(const float* __restrict__ x, float* __restrict__ outS,
                         float* __restrict__ ta) {
    int gid = blockIdx.x * blockDim.x + threadIdx.x;
    if (gid >= CHN) return;
    int b = gid / (NV * DM), r = gid % (NV * DM);
    int n = r >> 6, d = r & 63;
    float v[12], S = 0.0f;
    size_t idx0 = (((size_t)(b * LSEQ) * NV + n) << 6) + d;
    const size_t str = (size_t)NV * DM;
    for (int l = 0; l < 12; ++l) { v[l] = x[idx0 + l * str]; S += v[l]; }
    for (int l = 0; l < 12; ++l) {
        float m = (S + (float)(12 - l) * v[0] + (float)(l + 1) * v[11]) * 0.04f;
        size_t idx = idx0 + l * str;
        outS[idx] = v[l] - m;
        if (ta) ta[idx] += m;
    }
}

__global__ void k_trendmean(const float* __restrict__ x, float* __restrict__ tr) {
    int gid = blockIdx.x * blockDim.x + threadIdx.x;
    if (gid >= CHN) return;
    int b = gid / (NV * DM), r = gid % (NV * DM);
    int n = r >> 6, d = r & 63;
    size_t idx0 = (((size_t)(b * LSEQ) * NV + n) << 6) + d;
    const size_t str = (size_t)NV * DM;
    float s = 0.0f;
    for (int l = 0; l < 12; ++l) s += x[idx0 + l * str];
    tr[gid] = s * (1.0f / 12.0f);
}

// ========================= custom layernorm =================================
__global__ __launch_bounds__(64)
void k_ln1(const float* __restrict__ X, const float* __restrict__ w,
           const float* __restrict__ b, float* __restrict__ O) {
    __shared__ float red[64];
    int tok = blockIdx.x, d = threadIdx.x;
    float v = X[(size_t)tok * DM + d];
    red[d] = v; __syncthreads();
    for (int o = 32; o > 0; o >>= 1) { if (d < o) red[d] += red[d + o]; __syncthreads(); }
    float mu = red[0] * (1.0f / 64.0f);
    __syncthreads();
    float df = v - mu;
    red[d] = df * df; __syncthreads();
    for (int o = 32; o > 0; o >>= 1) { if (d < o) red[d] += red[d + o]; __syncthreads(); }
    float var = red[0] * (1.0f / 64.0f);
    O[(size_t)tok * DM + d] = df * rsqrtf(var + 1e-5f) * w[d] + b[d];
}

__global__ void k_ln2(float* __restrict__ buf) {  // subtract mean over L, in place
    int gid = blockIdx.x * blockDim.x + threadIdx.x;
    if (gid >= CHN) return;
    int b = gid / (NV * DM), r = gid % (NV * DM);
    int n = r >> 6, d = r & 63;
    size_t idx0 = (((size_t)(b * LSEQ) * NV + n) << 6) + d;
    const size_t str = (size_t)NV * DM;
    float v[12], s = 0.0f;
    for (int l = 0; l < 12; ++l) { v[l] = buf[idx0 + l * str]; s += v[l]; }
    float m = s * (1.0f / 12.0f);
    for (int l = 0; l < 12; ++l) buf[idx0 + l * str] = v[l] - m;
}

__global__ void k_deccomb(const float* __restrict__ sl, const float* __restrict__ tr,
                          const float* __restrict__ ta, float* __restrict__ o) {
    int gid = blockIdx.x * blockDim.x + threadIdx.x;
    if (gid >= TOK * DM) return;
    int d = gid & 63, tok = gid >> 6;
    int b = tok / (LSEQ * NV), rr = tok % (LSEQ * NV);
    int n = rr % NV;
    o[gid] = sl[gid] + tr[((size_t)(b * NV + n) << 6) + d] + ta[gid];
}

__global__ void k_finaldot(const float* __restrict__ a, const float* __restrict__ w,
                           const float* __restrict__ b, float* __restrict__ out) {
    int gid = blockIdx.x * blockDim.x + threadIdx.x;
    if (gid >= TOK) return;
    float s = b[0];
    for (int d = 0; d < DM; ++d) s += a[(size_t)gid * DM + d] * w[d];
    out[gid] = s;
}

// ============================= host driver ==================================
// Param indices assume jax pytree (sorted-dict) flatten of setup_inputs():
// X, TE, SE, then params leaves alphabetically (nested).
enum {
    IN_X = 0, IN_TE, IN_SE,
    DC_KB, DC_KW, DC_OB, DC_OW, DC_QB, DC_QW, DC_VB, DC_VW,   // dec0.cross
    D_FF1, D_FF2,                                             // dec0.ff1/ff2
    DS_KB, DS_KW, DS_OB, DS_OW, DS_QB, DS_QW, DS_VB, DS_VW,   // dec0.self
    DLN_B, DLN_W,
    EA_KB, EA_KW, EA_OB, EA_OW, EA_QB, EA_QW, EA_VB, EA_VW,   // enc0.attn
    E_FF1, E_FF2,
    ELN_B, ELN_W,
    G_O1B, G_O1W, G_O2B, G_O2W, G_XSW, G_XTB, G_XTW,          // gate
    LX_B, LX_W,
    O1_B, O1_W, O2_B, O2_W,
    SA_KB, SA_KW, SA_OB, SA_OW, SA_QB, SA_QW, SA_VB, SA_VW,   // sa
    S_S1B, S_S1W, S_S2B, S_S2W, S_T1B, S_T1W, S_T2B, S_T2W,   // ste
    N_PARAMS
};

extern "C" void kernel_launch(void* const* d_in, const int* in_sizes, int n_in,
                              void* d_out, int out_size, void* d_ws, size_t ws_size,
                              hipStream_t stream) {
    (void)in_sizes; (void)out_size;
    if (n_in < N_PARAMS || !d_ws) return;

    const float* Xin = (const float*)d_in[IN_X];
    const int*   TE  = (const int*)d_in[IN_TE];
    const float* SE  = (const float*)d_in[IN_SE];
    auto P = [&](int i) { return (const float*)d_in[i]; };

    const size_t U = (size_t)TOK * DM;  // 3,993,600 floats per slot
    if (ws_size < 13 * U * sizeof(float)) return;
    float* ws   = (float*)d_ws;
    float* bX   = ws + 0 * U;   // x embedding (persists)
    float* STEH = ws + 1 * U;
    float* STEP = ws + 2 * U;
    float* ENC  = ws + 3 * U;   // encoder output (persists)
    float* TA   = ws + 4 * U;   // decoder trend accumulator
    float* SM   = ws + 5 * U;   // small region
    float* se   = SM;                // 325*64
    float* te2  = SM + 32768;        // 384*64
    float* tr   = SM + 65536;        // 16*325*64
    float* tmp1 = SM + 1048576;      // small GEMM scratch
    float* T1 = ws + 6 * U;
    float* T2 = ws + 7 * U;
    float* T3 = ws + 8 * U;
    float* T4 = ws + 9 * U;
    float* T5 = ws + 10 * U;
    float* BIG = ws + 11 * U;   // 2 slots (TOK x 128)

    auto gemm = [&](const float* A, const float* W, const float* bias,
                    const float* add, float* out, int M, int N, int K, int act) {
        if (N == 64 && K == 64)
            launch_gemm_nk<64, 64>(A, W, bias, add, out, M, act, stream);
        else if (N == 64 && K == 128)
            launch_gemm_nk<64, 128>(A, W, bias, add, out, M, act, stream);
        else
            launch_gemm_nk<128, 64>(A, W, bias, add, out, M, act, stream);
    };
    const int EW = TOK * DM;                       // elementwise D-size
    auto G1 = [&](int n) { return dim3((n + 255) / 256); };

    // ---- embedding + ST embedding --------------------------------------
    k_embed<<<G1(EW), 256, 0, stream>>>(Xin, P(LX_W), P(LX_B), bX);
    gemm(SE, P(S_S1W), P(S_S1B), nullptr, tmp1, NV, 64, 64, 1);
    gemm(tmp1, P(S_S2W), P(S_S2B), nullptr, se, NV, 64, 64, 0);
    k_te1<<<G1(BN * 24 * DM), 256, 0, stream>>>(TE, P(S_T1W), P(S_T1B), tmp1);
    gemm(tmp1, P(S_T2W), P(S_T2B), nullptr, te2, BN * 24, 64, 64, 0);
    k_ste<<<G1(BN * 24 * NV * DM), 256, 0, stream>>>(se, te2, STEH, STEP);

    // ---- spatial attention branch (enc1) --------------------------------
    k_concat<<<G1(TOK * 128), 256, 0, stream>>>(bX, STEH, BIG);
    gemm(BIG, P(SA_QW), P(SA_QB), nullptr, T1, TOK, 64, 128, 1);
    gemm(BIG, P(SA_KW), P(SA_KB), nullptr, T2, TOK, 64, 128, 1);
    gemm(BIG, P(SA_VW), P(SA_VB), nullptr, T3, TOK, 64, 128, 1);
    k_sa_attn<<<dim3(BN * LSEQ * 8), dim3(352), 0, stream>>>(T1, T2, T3, T4);
    gemm(T4, P(SA_OW), P(SA_OB), nullptr, T5, TOK, 64, 64, 1);   // ENC1 = T5

    // ---- encoder layer (autocorr + ff + decomp) -------------------------
    k_add<<<G1(EW), 256, 0, stream>>>(bX, STEH, T1, EW);         // XE = x + ste_his
    gemm(T1, P(EA_QW), P(EA_QB), nullptr, T2, TOK, 64, 64, 0);
    gemm(T1, P(EA_KW), P(EA_KB), nullptr, T3, TOK, 64, 64, 0);
    gemm(T1, P(EA_VW), P(EA_VB), nullptr, T4, TOK, 64, 64, 0);
    k_ac_core<<<dim3(BN * NV), dim3(64), 0, stream>>>(T2, T3, T4, BIG);
    gemm(BIG, P(EA_OW), P(EA_OB), T1, T2, TOK, 64, 64, 0);       // + residual XE
    k_decomp<<<G1(CHN), 256, 0, stream>>>(T2, T3, nullptr);
    gemm(T3, P(E_FF1), nullptr, nullptr, BIG, TOK, 128, 64, 1);
    gemm(BIG, P(E_FF2), nullptr, T3, T2, TOK, 64, 128, 0);       // x + y
    k_decomp<<<G1(CHN), 256, 0, stream>>>(T2, T3, nullptr);      // h = T3
    k_ln1<<<dim3(TOK), dim3(64), 0, stream>>>(T3, P(ELN_W), P(ELN_B), T2);
    k_ln2<<<G1(CHN), 256, 0, stream>>>(T2);                      // ENC2 = T2

    // ---- gated fusion ---------------------------------------------------
    gemm(T5, P(G_XSW), nullptr, nullptr, T3, TOK, 64, 64, 0);
    gemm(T2, P(G_XTW), P(G_XTB), T3, T4, TOK, 64, 64, 2);        // gate = T4
    k_gatecomb<<<G1(EW), 256, 0, stream>>>(T4, T5, T2, T3);
    gemm(T3, P(G_O1W), P(G_O1B), nullptr, BIG, TOK, 128, 64, 1);
    gemm(BIG, P(G_O2W), P(G_O2B), nullptr, ENC, TOK, 64, 128, 0);

    // ---- decoder --------------------------------------------------------
    k_trendmean<<<G1(CHN), 256, 0, stream>>>(bX, tr);
    k_zero<<<G1(EW), 256, 0, stream>>>(TA, EW);
    // self autocorr on seasonal init = ste_pred
    gemm(STEP, P(DS_QW), P(DS_QB), nullptr, T1, TOK, 64, 64, 0);
    gemm(STEP, P(DS_KW), P(DS_KB), nullptr, T2, TOK, 64, 64, 0);
    gemm(STEP, P(DS_VW), P(DS_VB), nullptr, T3, TOK, 64, 64, 0);
    k_ac_core<<<dim3(BN * NV), dim3(64), 0, stream>>>(T1, T2, T3, BIG);
    gemm(BIG, P(DS_OW), P(DS_OB), STEP, T4, TOK, 64, 64, 0);     // X1
    k_decomp<<<G1(CHN), 256, 0, stream>>>(T4, T1, TA);           // X1s = T1
    // cross autocorr vs encoder output
    gemm(T1, P(DC_QW), P(DC_QB), nullptr, T2, TOK, 64, 64, 0);
    gemm(ENC, P(DC_KW), P(DC_KB), nullptr, T3, TOK, 64, 64, 0);
    gemm(ENC, P(DC_VW), P(DC_VB), nullptr, T4, TOK, 64, 64, 0);
    k_ac_core<<<dim3(BN * NV), dim3(64), 0, stream>>>(T2, T3, T4, BIG);
    gemm(BIG, P(DC_OW), P(DC_OB), T1, T2, TOK, 64, 64, 0);       // X2
    k_decomp<<<G1(CHN), 256, 0, stream>>>(T2, T3, TA);           // X2s = T3
    // feed-forward
    gemm(T3, P(D_FF1), nullptr, nullptr, BIG, TOK, 128, 64, 1);
    gemm(BIG, P(D_FF2), nullptr, T3, T1, TOK, 64, 128, 0);       // X3
    k_decomp<<<G1(CHN), 256, 0, stream>>>(T1, T2, TA);           // X3s = T2
    // layernorm + combine
    k_ln1<<<dim3(TOK), dim3(64), 0, stream>>>(T2, P(DLN_W), P(DLN_B), T1);
    k_ln2<<<G1(CHN), 256, 0, stream>>>(T1);                      // seasonal = T1
    k_deccomb<<<G1(EW), 256, 0, stream>>>(T1, tr, TA, T3);       // dec = T3
    // output head
    gemm(T3, P(O1_W), P(O1_B), nullptr, T4, TOK, 64, 64, 1);
    k_finaldot<<<G1(TOK), 256, 0, stream>>>(T4, P(O2_W), P(O2_B), (float*)d_out);
}
